// Model_28003186770671
// MI455X (gfx1250) — matrix operand order repaired and checked by reading, not compiled, not run on hardware
//
#include <hip/hip_runtime.h>

#define USERS   50000
#define ITEMS   20000
#define NTOT    70000
#define D       64
#define EMAIN   2000000
#define EMODAL  1000000
#define LAMBDA  0.2f
#define IMGDIM  4096
#define TXTDIM  768

typedef __attribute__((ext_vector_type(16))) __bf16 v16bf;
typedef __attribute__((ext_vector_type(8)))  __bf16 v8bf;
typedef __attribute__((ext_vector_type(4)))  __bf16 v4bf;
typedef __attribute__((ext_vector_type(8)))  float  v8f;

// ---------------- utility kernels (float4-vectorized) ----------------

__global__ void zero_kernel(float4* __restrict__ p, size_t n4) {
  size_t i = (size_t)blockIdx.x * blockDim.x + threadIdx.x;
  if (i < n4) p[i] = make_float4(0.f, 0.f, 0.f, 0.f);
}

// Convert W[K][64] f32 -> transposed bf16 WbT[64][K] (one-shot, tiny, L2-hot afterwards)
__global__ void cvt_wT_kernel(const float* __restrict__ W, __bf16* __restrict__ WbT, int K) {
  int i = blockIdx.x * blockDim.x + threadIdx.x;
  int total = 64 * K;
  if (i >= total) return;
  int n = i / K, k = i - n * K;
  WbT[(size_t)n * K + k] = (__bf16)W[(size_t)k * 64 + n];
}

// ---------------- WMMA projection GEMM: Out[M,64] = X[M,K] @ W[K,64] + bias ----------------
// 128 threads = 4 wave32; block owns a 16-row tile, each wave one 16-wide N chunk.
// A: f32 -> bf16, double-buffered LDS (one float4 load + one 8B LDS store per thread/K-step),
//    fragment = 2x ds_load_b128 (+shufflevector).
// B: read directly from bf16 transposed weights in global (two 16B chunks per lane at +0 and +32 B, L2-hot).
__launch_bounds__(128)
__global__ void proj_wmma_kernel(const float* __restrict__ X, const __bf16* __restrict__ WbT,
                                 const float* __restrict__ bias, float* __restrict__ Out,
                                 int K) {
  __shared__ __bf16 As[2][16][32];   // 2 KB, double buffered
  const int lane = threadIdx.x & 31;
  const int wave = threadIdx.x >> 5;
  const int m0   = blockIdx.x * 16;
  const int n0   = wave * 16;

  // A staging coords: thread t stages float4 at (row = t>>3, col = (t&7)*4)
  const int sr = threadIdx.x >> 3;        // 0..15
  const int sc = (threadIdx.x & 7) * 4;   // 0,4,...,28
  const float* Xrow = X + (size_t)(m0 + sr) * K + sc;

  // fragment coords (ISA 16-bit A 16x32 / B 32x16 layouts)
  const int am  = lane & 15;
  const int akg = (lane >> 4) * 8;    // K group for A: 0 or 8
  const int col = n0 + (lane & 15);
  const int bkg = (lane >> 4) * 8;    // K group for B: 0 or 8
  const __bf16* Bcol = WbT + (size_t)col * K;

  v8f acc = {};
  int p = 0;
  for (int k0 = 0; k0 < K; k0 += 32, p ^= 1) {
    // stage A tile (convert in flight), prefetch next tile
    float4 x = *(const float4*)(Xrow + k0);
    if (k0 + 32 < K) __builtin_prefetch(Xrow + k0 + 32, 0, 3);
    v4bf h;
    h[0] = (__bf16)x.x; h[1] = (__bf16)x.y; h[2] = (__bf16)x.z; h[3] = (__bf16)x.w;
    *(v4bf*)&As[p][sr][sc] = h;
    __syncthreads();

    // A fragment: two contiguous 16B LDS chunks
    v8bf alo = *(const v8bf*)&As[p][am][akg];
    v8bf ahi = *(const v8bf*)&As[p][am][16 + akg];
    v16bf a = __builtin_shufflevector(alo, ahi,
        0, 1, 2, 3, 4, 5, 6, 7, 8, 9, 10, 11, 12, 13, 14, 15);

    // B fragment: two 16B chunks (k = 8h.. and 16+8h..) of the transposed weight column
    const v8bf* bp = (const v8bf*)(Bcol + k0 + bkg);
    v8bf blo = bp[0], bhi = bp[2];
    v16bf b = __builtin_shufflevector(blo, bhi,
        0, 1, 2, 3, 4, 5, 6, 7, 8, 9, 10, 11, 12, 13, 14, 15);

    acc = __builtin_amdgcn_wmma_f32_16x16x32_bf16(
        /*neg_a=*/false, a, /*neg_b=*/false, b,
        /*c_mod=*/(short)0, acc, /*reuse_a=*/false, /*reuse_b=*/false);
  }

  // D layout: VGPR r holds (M = r (lanes 0-15) or r+8 (lanes 16-31), N = lane&15)
  const int rbase = (lane < 16) ? 0 : 8;
  const float bv  = bias[col];
  #pragma unroll
  for (int r = 0; r < 8; ++r) {
    Out[(size_t)(m0 + rbase + r) * 64 + col] = acc[r] + bv;
  }
}

// ---------------- row L2-normalize in place (wave per 64-wide row) ----------------
__global__ void l2norm_kernel(float* __restrict__ buf, int rows) {
  int wave = threadIdx.x >> 5, lane = threadIdx.x & 31;
  int row  = blockIdx.x * (blockDim.x >> 5) + wave;
  if (row >= rows) return;
  float2* p = (float2*)(buf + (size_t)row * D);
  float2 d  = p[lane];
  float ss  = d.x * d.x + d.y * d.y;
  #pragma unroll
  for (int o = 16; o >= 1; o >>= 1) ss += __shfl_xor(ss, o, 32);
  float s = 1.f / fmaxf(sqrtf(ss), 1e-12f);
  d.x *= s; d.y *= s;
  p[lane] = d;
}

// ---------------- COO SpMM: y[rows[e]] += vals[e] * concat(xu,xi)[cols[e]] ----------------
// Each wave owns a chunk of 32 edges: index/value loads are fully coalesced (one edge per
// lane), then the chunk is processed with __shfl broadcasts; per edge each lane moves a
// float2 of the 64-wide row (global_load_b64 gather + 2x global_atomic_add_f32 scatter).
// Dense operands (17.9 MB) are L2-resident on the 192 MB L2.
__global__ void spmm_kernel(const int* __restrict__ rows, const int* __restrict__ cols,
                            const float* __restrict__ vals,
                            const float* __restrict__ xu, const float* __restrict__ xi,
                            float* __restrict__ y, int E) {
  const int lane = threadIdx.x & 31;
  const int wid  = blockIdx.x * (blockDim.x >> 5) + (threadIdx.x >> 5);
  const int base = wid * 32;
  if (base >= E) return;
  const int cnt = min(32, E - base);

  int dst = 0, src = 0; float val = 0.f;
  if (lane < cnt) {
    dst = rows[base + lane];
    src = cols[base + lane];
    val = vals[base + lane];
  }
  for (int i = 0; i < cnt; ++i) {
    int   d = __shfl(dst, i, 32);
    int   s = __shfl(src, i, 32);
    float v = __shfl(val, i, 32);
    const float* xb = (s < USERS) ? (xu + (size_t)s * D)
                                  : (xi + (size_t)(s - USERS) * D);
    float2 t  = ((const float2*)xb)[lane];
    float* yp = y + (size_t)d * D + lane * 2;
    atomicAdd(yp + 0, v * t.x);
    atomicAdd(yp + 1, v * t.y);
  }
}

// ---------------- e = e1 + e2 + LAM*e_adj ; out[:, off:off+64] = e (float4) ----------------
__global__ void combine_store_kernel(const float4* __restrict__ e1, const float4* __restrict__ e2,
                                     const float4* __restrict__ eadj, float4* __restrict__ e,
                                     float4* __restrict__ out, int off4, size_t n4) {
  size_t i = (size_t)blockIdx.x * blockDim.x + threadIdx.x;
  if (i >= n4) return;
  float4 a = e1[i], b = e2[i], c = eadj[i];
  float4 v = make_float4(a.x + b.x + LAMBDA * c.x, a.y + b.y + LAMBDA * c.y,
                         a.z + b.z + LAMBDA * c.z, a.w + b.w + LAMBDA * c.w);
  e[i] = v;
  size_t row = i >> 4; int j = (int)(i & 15);
  out[row * 64 + off4 + j] = v;
}

// ---------------- out[:, off:off+64] += src (float4) ----------------
__global__ void acc_out_kernel(const float4* __restrict__ src, float4* __restrict__ out,
                               int off4, size_t n4) {
  size_t i = (size_t)blockIdx.x * blockDim.x + threadIdx.x;
  if (i >= n4) return;
  float4 s = src[i];
  size_t row = i >> 4; int j = (int)(i & 15);
  float4 o = out[row * 64 + off4 + j];
  o.x += s.x; o.y += s.y; o.z += s.z; o.w += s.w;
  out[row * 64 + off4 + j] = o;
}

// ---------------- host orchestration ----------------
extern "C" void kernel_launch(void* const* d_in, const int* in_sizes, int n_in,
                              void* d_out, int out_size, void* d_ws, size_t ws_size,
                              hipStream_t stream) {
  const float* image = (const float*)d_in[0];
  const float* text  = (const float*)d_in[1];
  const int*   ar    = (const int*)d_in[2];
  const int*   ac    = (const int*)d_in[3];
  const float* av    = (const float*)d_in[4];
  const int*   ir    = (const int*)d_in[5];
  const int*   ic    = (const int*)d_in[6];
  const float* iv    = (const float*)d_in[7];
  const int*   tr    = (const int*)d_in[8];
  const int*   tc    = (const int*)d_in[9];
  const float* tv    = (const float*)d_in[10];
  const float* u_sh  = (const float*)d_in[11];
  const float* i_sh  = (const float*)d_in[12];
  const float* Wi_sh = (const float*)d_in[13];
  const float* bi_sh = (const float*)d_in[14];
  const float* Wt_sh = (const float*)d_in[15];
  const float* bt_sh = (const float*)d_in[16];
  const float* u_vsp = (const float*)d_in[17];
  const float* i_vsp = (const float*)d_in[18];
  const float* Wi_vsp= (const float*)d_in[19];
  const float* bi_vsp= (const float*)d_in[20];
  const float* u_tsp = (const float*)d_in[21];
  const float* i_tsp = (const float*)d_in[22];
  const float* Wt_tsp= (const float*)d_in[23];
  const float* bt_tsp= (const float*)d_in[24];
  float* out = (float*)d_out;

  const size_t N64 = (size_t)NTOT * D;    // 4,480,000 floats
  const size_t I64 = (size_t)ITEMS * D;   // 1,280,000 floats
  const size_t N4  = N64 / 4;

  char* ws = (char*)d_ws;
  float* B0 = (float*)ws; ws += N64 * sizeof(float);
  float* B1 = (float*)ws; ws += N64 * sizeof(float);
  float* B2 = (float*)ws; ws += N64 * sizeof(float);
  float* B3 = (float*)ws; ws += N64 * sizeof(float);
  float* B4 = (float*)ws; ws += N64 * sizeof(float);
  float* F0 = (float*)ws; ws += I64 * sizeof(float);  // img_sh (normalized)
  float* F1 = (float*)ws; ws += I64 * sizeof(float);  // txt_sh
  float* F2 = (float*)ws; ws += I64 * sizeof(float);  // img_sp
  float* F3 = (float*)ws; ws += I64 * sizeof(float);  // txt_sp
  __bf16* WbI0 = (__bf16*)ws; ws += (size_t)IMGDIM * 64 * sizeof(__bf16);
  __bf16* WbT0 = (__bf16*)ws; ws += (size_t)TXTDIM * 64 * sizeof(__bf16);
  __bf16* WbI1 = (__bf16*)ws; ws += (size_t)IMGDIM * 64 * sizeof(__bf16);
  __bf16* WbT1 = (__bf16*)ws; ws += (size_t)TXTDIM * 64 * sizeof(__bf16);

  const int T = 256;
  auto blocks = [](size_t n, int t) { return (unsigned)((n + t - 1) / t); };

  // --- weight conversions to transposed bf16 ---
  cvt_wT_kernel<<<blocks((size_t)IMGDIM * 64, T), T, 0, stream>>>(Wi_sh,  WbI0, IMGDIM);
  cvt_wT_kernel<<<blocks((size_t)TXTDIM * 64, T), T, 0, stream>>>(Wt_sh,  WbT0, TXTDIM);
  cvt_wT_kernel<<<blocks((size_t)IMGDIM * 64, T), T, 0, stream>>>(Wi_vsp, WbI1, IMGDIM);
  cvt_wT_kernel<<<blocks((size_t)TXTDIM * 64, T), T, 0, stream>>>(Wt_tsp, WbT1, TXTDIM);

  // --- WMMA projections + row l2-normalize ---
  const unsigned projBlocks = ITEMS / 16;          // 1250, exact
  const unsigned normBlocks = (ITEMS + 7) / 8;
  proj_wmma_kernel<<<projBlocks, 128, 0, stream>>>(image, WbI0, bi_sh,  F0, IMGDIM);
  l2norm_kernel<<<normBlocks, 256, 0, stream>>>(F0, ITEMS);
  proj_wmma_kernel<<<projBlocks, 128, 0, stream>>>(text,  WbT0, bt_sh,  F1, TXTDIM);
  l2norm_kernel<<<normBlocks, 256, 0, stream>>>(F1, ITEMS);
  proj_wmma_kernel<<<projBlocks, 128, 0, stream>>>(image, WbI1, bi_vsp, F2, IMGDIM);
  l2norm_kernel<<<normBlocks, 256, 0, stream>>>(F2, ITEMS);
  proj_wmma_kernel<<<projBlocks, 128, 0, stream>>>(text,  WbT1, bt_tsp, F3, TXTDIM);
  l2norm_kernel<<<normBlocks, 256, 0, stream>>>(F3, ITEMS);

  const unsigned zb = blocks(N4, T);
  auto spmm = [&](const int* r, const int* c, const float* v, int E,
                  const float* xu, const float* xi, float* y) {
    zero_kernel<<<zb, T, 0, stream>>>((float4*)y, N4);
    unsigned waves = (unsigned)((E + 31) / 32);
    spmm_kernel<<<(waves + 7) / 8, 256, 0, stream>>>(r, c, v, xu, xi, y, E);
  };

  auto branch = [&](const float* uE, const float* iE, const float* featsN,
                    const int* mr, const int* mc, const float* mv, int outOff) {
    // e_adj = modal_graph @ concat(uE, iE)
    spmm(mr, mc, mv, EMODAL, uE, iE, B0);
    // e1 = adj @ concat(uE, l2norm(feats))
    spmm(ar, ac, av, EMAIN, uE, featsN, B1);
    // e2 = adj @ concat(e1[:USERS], iE)
    spmm(ar, ac, av, EMAIN, B1, iE, B2);
    // e = e1 + e2 + LAM*e_adj ; store into out slice
    combine_store_kernel<<<zb, T, 0, stream>>>((const float4*)B1, (const float4*)B2,
                                               (const float4*)B0, (float4*)B3,
                                               (float4*)out, outOff / 4, N4);
    // GNN layer 1: cur = adj @ e ; out += cur
    spmm(ar, ac, av, EMAIN, B3, B3 + (size_t)USERS * D, B4);
    acc_out_kernel<<<zb, T, 0, stream>>>((const float4*)B4, (float4*)out, outOff / 4, N4);
    // GNN layer 2: cur = adj @ cur ; out += cur
    spmm(ar, ac, av, EMAIN, B4, B4 + (size_t)USERS * D, B0);
    acc_out_kernel<<<zb, T, 0, stream>>>((const float4*)B0, (float4*)out, outOff / 4, N4);
  };

  branch(u_sh,  i_sh,  F0, ir, ic, iv, 0);    // ev_sh
  branch(u_sh,  i_sh,  F1, tr, tc, tv, 64);   // et_sh
  branch(u_vsp, i_vsp, F2, ir, ic, iv, 128);  // ev_sp
  branch(u_tsp, i_tsp, F3, tr, tc, tv, 192);  // et_sp
}
